// USER_14474039787934
// MI455X (gfx1250) — matrix operand order repaired and checked by reading, not compiled, not run on hardware
//
#include <hip/hip_runtime.h>

// ---------------------------------------------------------------------------
// Types
// ---------------------------------------------------------------------------
typedef __attribute__((ext_vector_type(16))) __bf16        v16bf;
typedef __attribute__((ext_vector_type(8)))  float         v8f;
typedef __attribute__((ext_vector_type(4)))  unsigned int  u32x4;
typedef __attribute__((ext_vector_type(4)))  float         f32x4;

union BFrag { v16bf v; u32x4 q[2]; };

#define NUSERS 4096
#define TSTEPS 64
#define SKEP   768
#define HDIM   256
#define KROLES 64
#define XS_LD  272   // 256 + 16 ushort pad (bank-conflict relief)
#define HB_LD  272

// Machine-scheduler fence: cap software-pipeline depth at the source-level
// lookahead inside one iteration.
#if defined(__has_builtin)
#if __has_builtin(__builtin_amdgcn_sched_barrier)
#define SCHED_FENCE() __builtin_amdgcn_sched_barrier(0)
#endif
#endif
#ifndef SCHED_FENCE
#define SCHED_FENCE()
#endif

static __device__ __forceinline__ float sigm(float x) {
    return 1.0f / (1.0f + __expf(-x));
}

static __device__ __forceinline__ unsigned short f2bf(float f) {
    union { float f; unsigned u; } v; v.f = f;
    unsigned r = v.u + 0x7FFFu + ((v.u >> 16) & 1u);   // round-to-nearest-even
    return (unsigned short)(r >> 16);
}

static __device__ __forceinline__ v8f v8f_zero() {
    v8f z = {0.f, 0.f, 0.f, 0.f, 0.f, 0.f, 0.f, 0.f};
    return z;
}

// A fragment (16x32 bf16): lane&15 = row M, lane>>4 picks K-half;
// per-lane one contiguous 16-elt run -> two ds_load_b128
static __device__ __forceinline__ v16bf load_a_lds(const unsigned short* base,
                                                   int ld, int kb, int lane) {
    const int m = lane & 15, khi = lane >> 4;
    const unsigned short* p = base + m * ld + kb + khi * 16;
    BFrag f;
    f.q[0] = *(const u32x4*)(p);
    f.q[1] = *(const u32x4*)(p + 8);
    return f.v;
}

// B fragment (32x16 bf16): lane n holds W row (col tile + n), K-contiguous
// -> two global_load_b128 (weights L2-resident). With the opaque zero
// byte-offset applied to the (still provably-global) base, row/kb fold into
// the 24-bit immediate offset field.
static __device__ __forceinline__ v16bf load_b_w(const unsigned short* W,
                                                 int row, int ldk, int kb, int khi) {
    const unsigned short* p = W + (size_t)row * ldk + kb + khi * 16;
    BFrag f;
    f.q[0] = ((const u32x4*)p)[0];
    f.q[1] = ((const u32x4*)p)[1];
    return f.v;
}

// ---------------------------------------------------------------------------
// K0: fp32 -> bf16 weight conversion
// ---------------------------------------------------------------------------
__global__ __launch_bounds__(256) void cvt_bf16_k(const float* __restrict__ src,
                                                  unsigned short* __restrict__ dst,
                                                  int n) {
    int i = blockIdx.x * 256 + threadIdx.x;
    if (i < n) dst[i] = f2bf(src[i]);
}

// ---------------------------------------------------------------------------
// K1: fused masked LSTM. 32 users / workgroup (128 WGs), 8 waves, each wave
// owns 32 hidden dims (8 gate-column tiles). WMMA bf16 with fp32 accum;
// c/h state register-resident; B fragments software-pipelined exactly one
// tile ahead. An opaque 64-bit ZERO OFFSET (refreshed through empty asm each
// timestep) is added to the weight bases: loads stay un-hoistable across t
// (no LICM/PRE -> no spills), per-tile offsets fold into immediates, and —
// unlike laundering the pointer itself — address-space inference still sees
// GLOBAL, so loads are global_load_b128 (LOADcnt only, no flat/DScnt coupling).
// ---------------------------------------------------------------------------
__global__ __launch_bounds__(256) void lstm_k(const float* __restrict__ hist,
                                              const float* __restrict__ b_ih,
                                              const float* __restrict__ b_hh,
                                              const int*   __restrict__ lengths,
                                              const unsigned short* __restrict__ Wih,  // [1024,768] bf16
                                              const unsigned short* __restrict__ Whh,  // [1024,256] bf16
                                              float* __restrict__ h_out)               // [N,256]
{
    __shared__ unsigned short xs[32 * XS_LD];   // x_t chunk, bf16
    __shared__ unsigned short hb[32 * HB_LD];   // h state, bf16 (WMMA A source)
    __shared__ int len_s[32];

    const int tid   = threadIdx.x;
    const int lane  = tid & 31;
    const int wave  = tid >> 5;
    const int laneN = lane & 15;
    const int laneH = lane >> 4;          // 0 / 1
    const int ub    = blockIdx.x * 32;    // user base
    const int wcb   = wave * 32;          // hidden-column base for this wave

    if (tid < 32) len_s[tid] = lengths[ub + tid];

    // tile ti (0..7) -> gate g = ti>>1, subtile s = ti&1; W row for this lane
    auto rowOf = [&](int ti) {
        return (ti >> 1) * HDIM + wcb + (ti & 1) * 16 + laneN;
    };

    float bias[4][2];
#pragma unroll
    for (int g = 0; g < 4; ++g)
#pragma unroll
        for (int s = 0; s < 2; ++s) {
            const int col = g * HDIM + wcb + s * 16 + laneN;
            bias[g][s] = b_ih[col] + b_hh[col];
        }

    v8f c_reg[2][2], h_reg[2][2];
#pragma unroll
    for (int sl = 0; sl < 2; ++sl)
#pragma unroll
        for (int s = 0; s < 2; ++s) { c_reg[sl][s] = v8f_zero(); h_reg[sl][s] = v8f_zero(); }

    // init h (bf16) = 0; each wave covers its own columns for all 32 users
#pragma unroll
    for (int sl = 0; sl < 2; ++sl)
#pragma unroll
        for (int s = 0; s < 2; ++s)
#pragma unroll
            for (int r = 0; r < 8; ++r) {
                const int u = sl * 16 + r + 8 * laneH;
                hb[u * HB_LD + wcb + s * 16 + laneN] = 0;
            }
    __syncthreads();

    // prime the B-fragment pipeline: first tile of (t=0, kc=0, kb=0)
    v16bf bc = load_b_w(Wih, rowOf(0), SKEP, 0, laneH);

    // Opaque zero element-offset, refreshed per timestep.
    unsigned long long zoff = 0;

    for (int t = 0; t < TSTEPS; ++t) {
        // Make the offset formally loop-variant: no pass can hoist/PRE the
        // weight loads out of the t loop (value is 0 at runtime). The base
        // pointers below stay derived from kernargs -> still global addrspace.
        asm volatile("" : "+v"(zoff));
        const unsigned short* WihT = Wih + zoff;
        const unsigned short* WhhT = Whh + zoff;

        v8f acc[2][4][2];
#pragma unroll
        for (int sl = 0; sl < 2; ++sl)
#pragma unroll
            for (int g = 0; g < 4; ++g)
#pragma unroll
                for (int s = 0; s < 2; ++s) acc[sl][g][s] = v8f_zero();

        // ---- input contribution: x_t @ W_ih^T, K staged in 3 chunks of 256
#pragma unroll
        for (int kc = 0; kc < 3; ++kc) {
            __syncthreads();   // previous readers of xs done
            for (int i = tid; i < 32 * 64; i += 256) {
                const int u  = i >> 6;
                const int dq = (i & 63) * 4;
                const float* src = hist + ((size_t)(ub + u) * TSTEPS + t) * SKEP + kc * 256 + dq;
                f32x4 f = *(const f32x4*)src;
                unsigned short* dst = &xs[u * XS_LD + dq];
                dst[0] = f2bf(f.x); dst[1] = f2bf(f.y);
                dst[2] = f2bf(f.z); dst[3] = f2bf(f.w);
                if (t + 1 < TSTEPS) __builtin_prefetch(src + SKEP, 0, 1);
            }
            __syncthreads();
            v16bf a0 = load_a_lds(&xs[0],          XS_LD, 0, lane);
            v16bf a1 = load_a_lds(&xs[16 * XS_LD], XS_LD, 0, lane);
#pragma unroll
            for (int kb = 0; kb < 256; kb += 32) {
#pragma unroll
                for (int ti = 0; ti < 8; ++ti) {
                    v16bf bn;
                    if (ti < 7)                     // next tile, same k
                        bn = load_b_w(WihT, rowOf(ti + 1), SKEP, kc * 256 + kb, laneH);
                    else if (kb < 224)              // first tile, next k-step
                        bn = load_b_w(WihT, rowOf(0), SKEP, kc * 256 + kb + 32, laneH);
                    else if (kc < 2)                // first tile, next K-chunk
                        bn = load_b_w(WihT, rowOf(0), SKEP, (kc + 1) * 256, laneH);
                    else                            // hand off to the W_hh phase
                        bn = load_b_w(WhhT, rowOf(0), HDIM, 0, laneH);
                    const int g = ti >> 1, s = ti & 1;
                    acc[0][g][s] = __builtin_amdgcn_wmma_f32_16x16x32_bf16(
                        false, a0, false, bc, (short)0, acc[0][g][s], false, false);
                    acc[1][g][s] = __builtin_amdgcn_wmma_f32_16x16x32_bf16(
                        false, a1, false, bc, (short)0, acc[1][g][s], false, false);
                    bc = bn;
                    SCHED_FENCE();                  // cap pipeline depth at 1 tile
                }
                if (kb < 224) {
                    a0 = load_a_lds(&xs[0],          XS_LD, kb + 32, lane);
                    a1 = load_a_lds(&xs[16 * XS_LD], XS_LD, kb + 32, lane);
                }
            }
        }

        // ---- recurrent contribution: h @ W_hh^T (h bf16 already in LDS)
        {
            v16bf a0 = load_a_lds(&hb[0],          HB_LD, 0, lane);
            v16bf a1 = load_a_lds(&hb[16 * HB_LD], HB_LD, 0, lane);
#pragma unroll
            for (int kb = 0; kb < 256; kb += 32) {
#pragma unroll
                for (int ti = 0; ti < 8; ++ti) {
                    v16bf bn;
                    if (ti < 7)
                        bn = load_b_w(WhhT, rowOf(ti + 1), HDIM, kb, laneH);
                    else if (kb < 224)
                        bn = load_b_w(WhhT, rowOf(0), HDIM, kb + 32, laneH);
                    else                            // weights are t-invariant:
                        bn = load_b_w(WihT, rowOf(0), SKEP, 0, laneH);  // preload t+1
                    const int g = ti >> 1, s = ti & 1;
                    acc[0][g][s] = __builtin_amdgcn_wmma_f32_16x16x32_bf16(
                        false, a0, false, bc, (short)0, acc[0][g][s], false, false);
                    acc[1][g][s] = __builtin_amdgcn_wmma_f32_16x16x32_bf16(
                        false, a1, false, bc, (short)0, acc[1][g][s], false, false);
                    bc = bn;
                    SCHED_FENCE();                  // cap pipeline depth at 1 tile
                }
                if (kb < 224) {
                    a0 = load_a_lds(&hb[0],          HB_LD, kb + 32, lane);
                    a1 = load_a_lds(&hb[16 * HB_LD], HB_LD, kb + 32, lane);
                }
            }
        }
        __syncthreads();   // all hb reads complete before rewriting h

        // ---- LSTM cell update, entirely in registers; D-tile layout:
        // acc[..][r] in lane l  <->  user (r + 8*laneH), hidden (laneN + tile)
#pragma unroll
        for (int sl = 0; sl < 2; ++sl)
#pragma unroll
            for (int s = 0; s < 2; ++s)
#pragma unroll
                for (int r = 0; r < 8; ++r) {
                    const int u  = sl * 16 + r + 8 * laneH;
                    const float iv = sigm (acc[sl][0][s][r] + bias[0][s]);
                    const float fv = sigm (acc[sl][1][s][r] + bias[1][s]);
                    const float gv = tanhf(acc[sl][2][s][r] + bias[2][s]);
                    const float ov = sigm (acc[sl][3][s][r] + bias[3][s]);
                    const float cold = c_reg[sl][s][r];
                    float cn = fv * cold + iv * gv;
                    float hn = ov * tanhf(cn);
                    const bool valid = t < len_s[u];
                    cn = valid ? cn : cold;
                    hn = valid ? hn : h_reg[sl][s][r];
                    c_reg[sl][s][r] = cn;
                    h_reg[sl][s][r] = hn;
                    hb[u * HB_LD + wcb + s * 16 + laneN] = f2bf(hn);
                }
        // next iteration's first __syncthreads orders hb writes vs reads
    }

    // h_n out (fp32)
#pragma unroll
    for (int sl = 0; sl < 2; ++sl)
#pragma unroll
        for (int s = 0; s < 2; ++s)
#pragma unroll
            for (int r = 0; r < 8; ++r) {
                const int u = sl * 16 + r + 8 * laneH;
                h_out[(size_t)(ub + u) * HDIM + wcb + s * 16 + laneN] = h_reg[sl][s][r];
            }
}

// ---------------------------------------------------------------------------
// K2a: s[n] = <cluster_center[cid[n]], user_vec[n]> / sqrt(H)
// ---------------------------------------------------------------------------
__global__ __launch_bounds__(256) void score_k(const float* __restrict__ uv,
                                               const float* __restrict__ cc,
                                               const int*   __restrict__ cid,
                                               float* __restrict__ sc, int n) {
    int i = blockIdx.x * 256 + threadIdx.x;
    if (i >= n) return;
    const float* u = uv + (size_t)i * HDIM;
    const float* c = cc + (size_t)cid[i] * HDIM;
    float acc = 0.f;
    for (int j = 0; j < HDIM; j += 4) {
        f32x4 a = *(const f32x4*)(u + j);
        f32x4 b = *(const f32x4*)(c + j);
        acc += a.x * b.x + a.y * b.y + a.z * b.z + a.w * b.w;
    }
    sc[i] = acc * 0.0625f;   // 1/sqrt(256)
}

// ---------------------------------------------------------------------------
// K2b: per-role stable segment softmax + weighted sum of h_n -> role_senti
// one block per role; membership gate is block-uniform so non-member rows skip
// ---------------------------------------------------------------------------
__global__ __launch_bounds__(256) void role_k(const float* __restrict__ sc,
                                              const int*   __restrict__ cid,
                                              const float* __restrict__ hn,
                                              float* __restrict__ role, int n) {
    __shared__ float red[256];
    __shared__ float e_s[NUSERS];
    const int k = blockIdx.x, tid = threadIdx.x;

    float mx = -1e30f;
    for (int i = tid; i < n; i += 256)
        if (cid[i] == k) mx = fmaxf(mx, sc[i]);
    red[tid] = mx; __syncthreads();
    for (int off = 128; off; off >>= 1) {
        if (tid < off) red[tid] = fmaxf(red[tid], red[tid + off]);
        __syncthreads();
    }
    const float smax = red[0]; __syncthreads();

    float sum = 0.f;
    for (int i = tid; i < n; i += 256) {
        float e = (cid[i] == k) ? __expf(sc[i] - smax) : 0.0f;
        e_s[i] = e; sum += e;
    }
    red[tid] = sum; __syncthreads();
    for (int off = 128; off; off >>= 1) {
        if (tid < off) red[tid] += red[tid + off];
        __syncthreads();
    }
    const float den = red[0]; __syncthreads();

    float acc = 0.f;               // thread tid owns hidden dim `tid`
    for (int i = 0; i < n; ++i) {
        const float e = e_s[i];    // uniform across block -> cheap skip
        if (e != 0.0f) acc += e * hn[(size_t)i * HDIM + tid];
    }
    role[(size_t)k * HDIM + tid] = acc / den;
}

// ---------------------------------------------------------------------------
// K3a: q/k/v projections, one block per role
// ---------------------------------------------------------------------------
__global__ __launch_bounds__(256) void qkv_k(const float* __restrict__ role,
                                             const float* __restrict__ Wq, const float* __restrict__ bq,
                                             const float* __restrict__ Wk, const float* __restrict__ bk,
                                             const float* __restrict__ Wv, const float* __restrict__ bv,
                                             float* __restrict__ q, float* __restrict__ kk,
                                             float* __restrict__ v) {
    __shared__ float x[HDIM];
    const int i = blockIdx.x, j = threadIdx.x;
    x[j] = role[i * HDIM + j];
    __syncthreads();
    float aq = 0.f, ak = 0.f, av = 0.f;
    for (int d = 0; d < HDIM; ++d) {
        const float xv = x[d];
        aq += xv * Wq[j * HDIM + d];
        ak += xv * Wk[j * HDIM + d];
        av += xv * Wv[j * HDIM + d];
    }
    q [i * HDIM + j] = aq + bq[j];
    kk[i * HDIM + j] = ak + bk[j];
    v [i * HDIM + j] = av + bv[j];
}

// ---------------------------------------------------------------------------
// K3b: per-head attention over 64 roles (one block per head)
// ---------------------------------------------------------------------------
__global__ __launch_bounds__(256) void attn_k(const float* __restrict__ q,
                                              const float* __restrict__ k,
                                              const float* __restrict__ v,
                                              float* __restrict__ out) {
    __shared__ float scs[KROLES * KROLES];
    const int h = blockIdx.x, tid = threadIdx.x;

    for (int p = tid; p < KROLES * KROLES; p += 256) {
        const int qi = p >> 6, ki = p & 63;
        const float* qp = q + qi * HDIM + h * 32;
        const float* kp = k + ki * HDIM + h * 32;
        float a = 0.f;
        for (int d = 0; d < 32; ++d) a += qp[d] * kp[d];
        scs[p] = a * 0.17677669529663687f;   // 1/sqrt(32)
    }
    __syncthreads();
    if (tid < KROLES) {
        float mx = -1e30f;
        for (int j = 0; j < KROLES; ++j) mx = fmaxf(mx, scs[tid * KROLES + j]);
        float s = 0.f;
        for (int j = 0; j < KROLES; ++j) {
            float e = __expf(scs[tid * KROLES + j] - mx);
            scs[tid * KROLES + j] = e; s += e;
        }
        const float inv = 1.0f / s;
        for (int j = 0; j < KROLES; ++j) scs[tid * KROLES + j] *= inv;
    }
    __syncthreads();
    for (int p = tid; p < KROLES * 32; p += 256) {
        const int qi = p >> 5, d = p & 31;
        float a = 0.f;
        for (int j = 0; j < KROLES; ++j)
            a += scs[qi * KROLES + j] * v[j * HDIM + h * 32 + d];
        out[qi * HDIM + h * 32 + d] = a;
    }
}

// ---------------------------------------------------------------------------
// K3c: mean over roles commutes with output projection:
//      public = mean_i(att_out[i]) @ Wo^T + bo
// ---------------------------------------------------------------------------
__global__ __launch_bounds__(256) void outproj_k(const float* __restrict__ ao,
                                                 const float* __restrict__ Wo,
                                                 const float* __restrict__ bo,
                                                 float* __restrict__ out) {
    __shared__ float m[HDIM];
    const int j = threadIdx.x;
    float s = 0.f;
    for (int i = 0; i < KROLES; ++i) s += ao[i * HDIM + j];
    m[j] = s * (1.0f / (float)KROLES);
    __syncthreads();
    float a = 0.f;
    for (int d = 0; d < HDIM; ++d) a += m[d] * Wo[j * HDIM + d];
    out[j] = a + bo[j];
}

// ---------------------------------------------------------------------------
extern "C" void kernel_launch(void* const* d_in, const int* in_sizes, int n_in,
                              void* d_out, int out_size, void* d_ws, size_t ws_size,
                              hipStream_t stream) {
    const float* hist    = (const float*)d_in[0];
    const float* uv      = (const float*)d_in[1];
    const float* cc      = (const float*)d_in[2];
    const float* W_ih    = (const float*)d_in[3];
    const float* W_hh    = (const float*)d_in[4];
    const float* b_ih    = (const float*)d_in[5];
    const float* b_hh    = (const float*)d_in[6];
    const float* Wq      = (const float*)d_in[7];
    const float* bq      = (const float*)d_in[8];
    const float* Wk      = (const float*)d_in[9];
    const float* bk      = (const float*)d_in[10];
    const float* Wv      = (const float*)d_in[11];
    const float* bv      = (const float*)d_in[12];
    const float* Wo      = (const float*)d_in[13];
    const float* bo      = (const float*)d_in[14];
    const int*   lengths = (const int*)d_in[15];
    const int*   cid     = (const int*)d_in[16];

    char* ws = (char*)d_ws;
    unsigned short* Wih_bf = (unsigned short*)(ws + 0);        // 1,572,864 B
    unsigned short* Whh_bf = (unsigned short*)(ws + 1572864);  //   524,288 B
    float* hn   = (float*)(ws + 2097152);                      // 4,194,304 B
    float* sc   = (float*)(ws + 6291456);                      //    16,384 B
    float* role = (float*)(ws + 6307840);                      //    65,536 B
    float* qb   = (float*)(ws + 6373376);                      //    65,536 B
    float* kbuf = qb + KROLES * HDIM;
    float* vbuf = kbuf + KROLES * HDIM;
    float* ao   = (float*)(ws + 6569984);                      //    65,536 B

    cvt_bf16_k<<<(1024 * SKEP + 255) / 256, 256, 0, stream>>>(W_ih, Wih_bf, 1024 * SKEP);
    cvt_bf16_k<<<(1024 * HDIM + 255) / 256, 256, 0, stream>>>(W_hh, Whh_bf, 1024 * HDIM);
    lstm_k<<<NUSERS / 32, 256, 0, stream>>>(hist, b_ih, b_hh, lengths, Wih_bf, Whh_bf, hn);
    score_k<<<NUSERS / 256, 256, 0, stream>>>(uv, cc, cid, sc, NUSERS);
    role_k<<<KROLES, 256, 0, stream>>>(sc, cid, hn, role, NUSERS);
    qkv_k<<<KROLES, 256, 0, stream>>>(role, Wq, bq, Wk, bk, Wv, bv, qb, kbuf, vbuf);
    attn_k<<<8, 256, 0, stream>>>(qb, kbuf, vbuf, ao);
    outproj_k<<<1, 256, 0, stream>>>(ao, Wo, bo, (float*)d_out);
}
